// ProcessorExp_49804440764999
// MI455X (gfx1250) — compile-verified, matrix operand order checked by reading, not run on hardware
//
#include <hip/hip_runtime.h>
#include <hip/hip_bf16.h>
#include <cstdint>
#include <cstddef>

typedef __attribute__((ext_vector_type(16))) _Float16 v16h;
typedef __attribute__((ext_vector_type(2)))  _Float16 v2h;
typedef __attribute__((ext_vector_type(8)))  float    v8f;

#define B_    8
#define N_    4096
#define C1_   64
#define NP0_  512
#define NP1_  256
#define NS_   32
#define D0_   128
#define D1_   256
#define EPS_  1e-5f

// ---------------------------------------------------------------------------
// Stage 1: pointwise convs + global batchnorm
// ---------------------------------------------------------------------------

__global__ void __launch_bounds__(256)
conv1_kernel(const float* __restrict__ x, const float* __restrict__ w,
             float* __restrict__ y1)
{
    const int n = blockIdx.x * 256 + threadIdx.x;
    const int o = blockIdx.y;
    const int b = blockIdx.z;
    if (n >= N_) return;
    const float x0 = x[((size_t)b * 3 + 0) * N_ + n];
    const float x1 = x[((size_t)b * 3 + 1) * N_ + n];
    const float x2 = x[((size_t)b * 3 + 2) * N_ + n];
    y1[((size_t)b * C1_ + o) * N_ + n] =
        x0 * w[o * 3 + 0] + x1 * w[o * 3 + 1] + x2 * w[o * 3 + 2];
}

__global__ void __launch_bounds__(256)
conv_stats_kernel(const float* __restrict__ y, const float* __restrict__ g,
                  const float* __restrict__ beta, float* __restrict__ sb,
                  int C, int Nn, int Bb)
{
    const int ch = blockIdx.x;
    const int tid = threadIdx.x;
    __shared__ float s1[256];
    __shared__ float s2[256];
    float a = 0.f, q = 0.f;
    const int total = Bb * Nn;
    for (int e = tid; e < total; e += 256) {
        const int bb = e / Nn, n = e - bb * Nn;
        const float v = y[((size_t)bb * C + ch) * Nn + n];
        a += v; q += v * v;
    }
    s1[tid] = a; s2[tid] = q;
    __syncthreads();
    for (int st = 128; st > 0; st >>= 1) {
        if (tid < st) { s1[tid] += s1[tid + st]; s2[tid] += s2[tid + st]; }
        __syncthreads();
    }
    if (tid == 0) {
        const float m   = s1[0] / (float)total;
        const float var = s2[0] / (float)total - m * m;
        const float sc  = g[ch] * rsqrtf(var + EPS_);
        sb[ch]     = sc;
        sb[C + ch] = beta[ch] - m * sc;
    }
}

__global__ void __launch_bounds__(256)
bnrelu_kernel(const float* __restrict__ y, const float* __restrict__ sb,
              float* __restrict__ out, int total)
{
    const int idx = blockIdx.x * 256 + threadIdx.x;
    if (idx >= total) return;
    const int c = (idx / N_) % C1_;
    out[idx] = fmaxf(y[idx] * sb[c] + sb[C1_ + c], 0.f);
}

__global__ void __launch_bounds__(256)
conv2_kernel(const float* __restrict__ y1n, const float* __restrict__ w2,
             float* __restrict__ y2)
{
    const int n = blockIdx.x * 256 + threadIdx.x;
    const int o = blockIdx.y;
    const int b = blockIdx.z;
    if (n >= N_) return;
    float acc = 0.f;
    const float* yb = y1n + (size_t)b * C1_ * N_ + n;
    const float* wo = w2 + o * C1_;
#pragma unroll 8
    for (int c = 0; c < C1_; ++c)
        acc += yb[(size_t)c * N_] * wo[c];
    y2[((size_t)b * C1_ + o) * N_ + n] = acc;
}

__global__ void __launch_bounds__(256)
pts_kernel(const float* __restrict__ y2, const float* __restrict__ sb,
           _Float16* __restrict__ ptsh, int total)
{
    const int idx = blockIdx.x * 256 + threadIdx.x;
    if (idx >= total) return;
    const int c  = idx & (C1_ - 1);
    const int bn = idx >> 6;
    const int n  = bn & (N_ - 1);
    const int b  = bn >> 12;
    const float v = y2[((size_t)b * C1_ + c) * N_ + n];
    ptsh[idx] = (_Float16)fmaxf(v * sb[c] + sb[C1_ + c], 0.f);
}

__global__ void __launch_bounds__(256)
xyz_kernel(const float* __restrict__ x, float* __restrict__ xyzw)
{
    const int idx = blockIdx.x * 256 + threadIdx.x;   // over B*N
    if (idx >= B_ * N_) return;
    const int n = idx & (N_ - 1);
    const int b = idx >> 12;
    xyzw[(size_t)idx * 4 + 0] = x[((size_t)b * 3 + 0) * N_ + n];
    xyzw[(size_t)idx * 4 + 1] = x[((size_t)b * 3 + 1) * N_ + n];
    xyzw[(size_t)idx * 4 + 2] = x[((size_t)b * 3 + 2) * N_ + n];
    xyzw[(size_t)idx * 4 + 3] = 0.f;
}

// ---------------------------------------------------------------------------
// Stage 2: FPS + kNN (latency-bound LDS reduction kernels)
// ---------------------------------------------------------------------------

template<int NPTS, int NOUT, int TPB>
__global__ void __launch_bounds__(TPB)
fps_kernel(const float* __restrict__ xyz /*(B,NPTS,4)*/, int* __restrict__ outIdx)
{
    const int b = blockIdx.x;
    const int tid = threadIdx.x;
    __shared__ float dmin[NPTS];
    __shared__ float rv[TPB];
    __shared__ int   ri[TPB];
    __shared__ float cen[3];
    __shared__ int   cur;
    for (int i = tid; i < NPTS; i += TPB) dmin[i] = 1e10f;
    if (tid == 0) cur = 0;
    __syncthreads();
    const float* base = xyz + (size_t)b * NPTS * 4;
    for (int it = 0; it < NOUT; ++it) {
        if (tid == 0) {
            const int f = cur;
            outIdx[b * NOUT + it] = f;
            cen[0] = base[f * 4 + 0];
            cen[1] = base[f * 4 + 1];
            cen[2] = base[f * 4 + 2];
        }
        __syncthreads();
        const float cx = cen[0], cy = cen[1], cz = cen[2];
        float best = -1.f; int bi = 0;
        for (int i = tid; i < NPTS; i += TPB) {
            const float dx = base[i * 4 + 0] - cx;
            const float dy = base[i * 4 + 1] - cy;
            const float dz = base[i * 4 + 2] - cz;
            const float d  = dx * dx + dy * dy + dz * dz;
            const float nd = fminf(dmin[i], d);
            dmin[i] = nd;
            if (nd > best) { best = nd; bi = i; }
        }
        rv[tid] = best; ri[tid] = bi;
        __syncthreads();
        for (int st = TPB / 2; st > 0; st >>= 1) {
            if (tid < st) {
                if (rv[tid + st] > rv[tid] ||
                    (rv[tid + st] == rv[tid] && ri[tid + st] < ri[tid])) {
                    rv[tid] = rv[tid + st]; ri[tid] = ri[tid + st];
                }
            }
            __syncthreads();
        }
        if (tid == 0) cur = ri[0];
        __syncthreads();
    }
}

__global__ void __launch_bounds__(256)
gatherxyz_kernel(const float* __restrict__ src4, const int* __restrict__ idx,
                 float* __restrict__ dst4, int np, int srcN, int count)
{
    const int i = blockIdx.x * 256 + threadIdx.x;
    if (i >= count) return;
    const int b = i / np;
    const int f = idx[i];
    const float* s = src4 + ((size_t)b * srcN + f) * 4;
    dst4[(size_t)i * 4 + 0] = s[0];
    dst4[(size_t)i * 4 + 1] = s[1];
    dst4[(size_t)i * 4 + 2] = s[2];
    dst4[(size_t)i * 4 + 3] = 0.f;
}

template<int NSRC, int TPB>
__global__ void __launch_bounds__(TPB)
knn_kernel(const float* __restrict__ q /*(B*NQ,4)*/,
           const float* __restrict__ src /*(B,NSRC,4)*/,
           int* __restrict__ knn, int NQ)
{
    const int qi = blockIdx.x;
    const int b  = qi / NQ;
    const int tid = threadIdx.x;
    __shared__ float dist[NSRC];
    __shared__ float rv[TPB];
    __shared__ int   ri[TPB];
    const float qx = q[(size_t)qi * 4 + 0];
    const float qy = q[(size_t)qi * 4 + 1];
    const float qz = q[(size_t)qi * 4 + 2];
    const float* sp = src + (size_t)b * NSRC * 4;
    for (int i = tid; i < NSRC; i += TPB) {
        const float dx = sp[i * 4 + 0] - qx;
        const float dy = sp[i * 4 + 1] - qy;
        const float dz = sp[i * 4 + 2] - qz;
        dist[i] = dx * dx + dy * dy + dz * dz;
    }
    __syncthreads();
    for (int k = 0; k < NS_; ++k) {
        float best = 3.5e38f; int bi = 0x7fffffff;
        for (int i = tid; i < NSRC; i += TPB) {
            const float d = dist[i];
            if (d < best || (d == best && i < bi)) { best = d; bi = i; }
        }
        rv[tid] = best; ri[tid] = bi;
        __syncthreads();
        for (int st = TPB / 2; st > 0; st >>= 1) {
            if (tid < st) {
                if (rv[tid + st] < rv[tid] ||
                    (rv[tid + st] == rv[tid] && ri[tid + st] < ri[tid])) {
                    rv[tid] = rv[tid + st]; ri[tid] = ri[tid + st];
                }
            }
            __syncthreads();
        }
        if (tid == 0) { knn[(size_t)qi * NS_ + k] = ri[0]; dist[ri[0]] = 3.5e38f; }
        __syncthreads();
    }
}

// ---------------------------------------------------------------------------
// Stage 3: local-encoder WMMA GEMMs with pre-packed LDS fragments
//   out(grp, o, j) = sum_k W(o,k) * Ft(k, j);  M=DF, N=32, K=DF
//   Fragment order in LDS: [tile][lane][16 halves] -> one v16h load per operand.
//   A pack: lane L<16 row M=L holds K {0..7,16..23}; L>=16 -> {8..15,24..31};
//   B pack mirrors with N on lanes. Inverse map used on staging stores:
//     laneSel = ((k>>3)&1)*16 + m_or_n;  h = (k&7) | ((k>>4)<<3)
// ---------------------------------------------------------------------------

__device__ __forceinline__ int bfrag_idx(int k, int n)
{
    const int kk = k >> 5, kl = k & 31, t = n >> 4;
    const int laneD = ((kl >> 3) & 1) * 16 + (n & 15);
    const int h = (kl & 7) | ((kl >> 4) << 3);
    return ((kk * 2 + t) * 32 + laneD) * 16 + h;
}

template<int DF>
__device__ __forceinline__ void wmma_block_gemm(
    const _Float16* __restrict__ W,
    const _Float16* FtP,            // packed B fragments, DF*32 halves
    _Float16* WaP,                  // packed A staging, DF*32 halves
    _Float16* __restrict__ out, int grp)
{
    constexpr int KSTEPS = DF / 32;
    constexpr int RT     = DF / 128;     // 16-row stripes per wave
    constexpr int ITER   = (DF * 4) / 256;  // async 16B chunks per thread (2 or 4)
    const int tid  = threadIdx.x;
    const int lane = tid & 31;
    const int wave = tid >> 5;
    const int lm   = lane & 15;
    const int lh   = lane >> 4;

    const v16h* FtV = (const v16h*)FtP;
    const v16h* WaV = (const v16h*)WaP;
    const unsigned long long wbase = (unsigned long long)W;

    v8f acc[RT][2];
#pragma unroll
    for (int r = 0; r < RT; ++r)
#pragma unroll
        for (int t = 0; t < 2; ++t) {
            v8f z = {0.f, 0.f, 0.f, 0.f, 0.f, 0.f, 0.f, 0.f};
            acc[r][t] = z;
        }

    for (int kk = 0; kk < KSTEPS; ++kk) {
        __syncthreads();
        // Async-stage W[:, kk*32 .. +32) straight into packed fragment order.
        // 16B global chunk (8 consecutive k of one row) lands contiguously in
        // LDS. Trip count is exact -> straight-line clause of async issues.
#pragma unroll
        for (int i = 0; i < ITER; ++i) {
            const int c8 = tid + i * 256;
            const int r  = c8 >> 2;          // weight row (output channel)
            const int c0 = (c8 & 3) * 8;     // k offset within slab, multiple of 8
            const int laneD = (r & 15) + (((c0 >> 3) & 1) << 4);
            const int h0    = (c0 >> 4) * 8;
            const int rt    = r >> 4;
            const unsigned ldsa = (unsigned)(uintptr_t)&WaP[(rt * 32 + laneD) * 16 + h0];
            const unsigned voff = (unsigned)(((size_t)r * DF + (kk << 5) + c0) * 2);
            asm volatile("global_load_async_to_lds_b128 %0, %1, %2"
                         :: "v"(ldsa), "v"(voff), "s"(wbase) : "memory");
        }
        asm volatile("s_wait_asynccnt 0x0" ::: "memory");
        __syncthreads();
        if (kk + 1 < KSTEPS)
            __builtin_prefetch(W + (size_t)(tid & (DF - 1)) * DF + ((kk + 1) << 5), 0, 0);
#pragma unroll
        for (int r = 0; r < RT; ++r) {
            const int rt = wave * RT + r;
            const v16h a = WaV[rt * 32 + lane];
#pragma unroll
            for (int t = 0; t < 2; ++t) {
                const v16h bm = FtV[(kk * 2 + t) * 32 + lane];
                acc[r][t] = __builtin_amdgcn_wmma_f32_16x16x32_f16(
                    false, a, false, bm, (short)0, acc[r][t], false, false);
            }
        }
    }

    _Float16* ob = out + (size_t)grp * DF * 32;
#pragma unroll
    for (int r = 0; r < RT; ++r) {
        const int rt = wave * RT + r;
#pragma unroll
        for (int t = 0; t < 2; ++t)
#pragma unroll
            for (int p = 0; p < 8; ++p) {
                const int m = rt * 16 + (lh << 3) + p;
                const int n = t * 16 + lm;
                ob[(size_t)m * 32 + n] = (_Float16)acc[r][t][p];
            }
    }
}

// First encoder matmul: build feat = [nb - ctr ; ctr] packed in LDS, then GEMM.
template<int DF>
__global__ void __launch_bounds__(256)
enc_gemm1_kernel(const _Float16* __restrict__ pts, int nsrc,
                 const int* __restrict__ fpsIdx,
                 const int* __restrict__ knnIdx,
                 const _Float16* __restrict__ W,
                 _Float16* __restrict__ out, int np)
{
    constexpr int DH  = DF / 2;
    constexpr int DH2 = DH / 2;
    __shared__ __align__(32) _Float16 FtP[DF * 32];
    __shared__ __align__(32) _Float16 WaP[DF * 32];
    __shared__ _Float16 ctr[DH];
    __shared__ int sk[NS_];
    const int grp = blockIdx.x;           // b*np + s
    const int b   = grp / np;
    const int tid = threadIdx.x;

    const _Float16* cp = pts + ((size_t)b * nsrc + fpsIdx[grp]) * DH;
    for (int c = tid; c < DH; c += 256) ctr[c] = cp[c];
    if (tid < NS_) sk[tid] = knnIdx[(size_t)grp * NS_ + tid];
    __syncthreads();

    for (int e2 = tid; e2 < NS_ * DH2; e2 += 256) {
        const int j  = e2 / DH2;
        const int c0 = (e2 - j * DH2) * 2;
        const v2h nb = *(const v2h*)&pts[((size_t)b * nsrc + sk[j]) * DH + c0];
        const _Float16 cv0 = ctr[c0], cv1 = ctr[c0 + 1];
        FtP[bfrag_idx(c0,          j)] = (_Float16)((float)nb.x - (float)cv0);
        FtP[bfrag_idx(c0 + 1,      j)] = (_Float16)((float)nb.y - (float)cv1);
        FtP[bfrag_idx(DH + c0,     j)] = cv0;
        FtP[bfrag_idx(DH + c0 + 1, j)] = cv1;
    }
    __syncthreads();
    wmma_block_gemm<DF>(W, FtP, WaP, out, grp);
}

// Second encoder matmul: BN+ReLU on load, pack, GEMM.
template<int DF>
__global__ void __launch_bounds__(256)
enc_gemm2_kernel(const _Float16* __restrict__ in, const float* __restrict__ sb,
                 const _Float16* __restrict__ W, _Float16* __restrict__ out)
{
    __shared__ __align__(32) _Float16 FtP[DF * 32];
    __shared__ __align__(32) _Float16 WaP[DF * 32];
    const int grp = blockIdx.x;
    const int tid = threadIdx.x;
    const v2h* in2 = (const v2h*)(in + (size_t)grp * DF * 32);
    for (int e2 = tid; e2 < DF * 16; e2 += 256) {
        const int k  = e2 >> 4;
        const int j0 = (e2 & 15) << 1;
        const v2h w = in2[e2];
        const float sc = sb[k], bi = sb[DF + k];
        FtP[bfrag_idx(k, j0)]     = (_Float16)fmaxf((float)w.x * sc + bi, 0.f);
        FtP[bfrag_idx(k, j0 + 1)] = (_Float16)fmaxf((float)w.y * sc + bi, 0.f);
    }
    __syncthreads();
    wmma_block_gemm<DF>(W, FtP, WaP, out, grp);
}

// per-channel stats over (grp, j) for layout ((grp*DF + ch)*32 + j)
__global__ void __launch_bounds__(256)
enc_stats_kernel(const _Float16* __restrict__ buf, int DF, int nG,
                 const float* __restrict__ g, const float* __restrict__ beta,
                 float* __restrict__ sb)
{
    const int ch = blockIdx.x;
    const int tid = threadIdx.x;
    __shared__ float s1[256];
    __shared__ float s2[256];
    float a = 0.f, q = 0.f;
    const int total2 = nG * 16;           // v2h pairs per channel
    for (int e = tid; e < total2; e += 256) {
        const int grp = e >> 4, jp = e & 15;
        const v2h w = *(const v2h*)&buf[((size_t)grp * DF + ch) * 32 + jp * 2];
        const float v0 = (float)w.x, v1 = (float)w.y;
        a += v0 + v1; q += v0 * v0 + v1 * v1;
    }
    s1[tid] = a; s2[tid] = q;
    __syncthreads();
    for (int st = 128; st > 0; st >>= 1) {
        if (tid < st) { s1[tid] += s1[tid + st]; s2[tid] += s2[tid + st]; }
        __syncthreads();
    }
    if (tid == 0) {
        const float tot = (float)(nG * NS_);
        const float m   = s1[0] / tot;
        const float var = s2[0] / tot - m * m;
        const float sc  = g[ch] * rsqrtf(var + EPS_);
        sb[ch]      = sc;
        sb[DF + ch] = beta[ch] - m * sc;
    }
}

// BN+ReLU then max over the 32 neighbors. TO_OUT: f32 (b,o,n); else f16 pts.
template<int DF, bool TO_OUT>
__global__ void __launch_bounds__(256)
enc_final_kernel(const _Float16* __restrict__ buf, const float* __restrict__ sb,
                 _Float16* __restrict__ ph, float* __restrict__ fo,
                 int np, int total)
{
    const int idx = blockIdx.x * 256 + threadIdx.x;  // over B*np*DF
    if (idx >= total) return;
    const int o   = idx % DF;
    const int grp = idx / DF;
    const float sc = sb[o], bi = sb[DF + o];
    const v16h* pv = (const v16h*)(buf + ((size_t)grp * DF + o) * 32);
    float mx = 0.f;
#pragma unroll
    for (int h2 = 0; h2 < 2; ++h2) {
        const v16h w = pv[h2];
#pragma unroll
        for (int j = 0; j < 16; ++j) {
            const float v = fmaxf((float)w[j] * sc + bi, 0.f);
            mx = fmaxf(mx, v);
        }
    }
    if (TO_OUT) {
        const int b = grp / np, s = grp - b * np;
        fo[((size_t)b * DF + o) * np + s] = mx;
    } else {
        ph[(size_t)grp * DF + o] = (_Float16)mx;
    }
}

__global__ void __launch_bounds__(256)
cvt_kernel(const float* __restrict__ src, _Float16* __restrict__ dst, int n)
{
    const int i = blockIdx.x * 256 + threadIdx.x;
    if (i < n) dst[i] = (_Float16)src[i];
}

// ---------------------------------------------------------------------------
// Launch
// ---------------------------------------------------------------------------

extern "C" void kernel_launch(void* const* d_in, const int* in_sizes, int n_in,
                              void* d_out, int out_size, void* d_ws, size_t ws_size,
                              hipStream_t stream)
{
    (void)in_sizes; (void)n_in; (void)out_size; (void)ws_size;
    const float* x       = (const float*)d_in[0];
    const float* conv1_w = (const float*)d_in[1];
    const float* bn1_g   = (const float*)d_in[2];
    const float* bn1_b   = (const float*)d_in[3];
    const float* conv2_w = (const float*)d_in[4];
    const float* bn2_g   = (const float*)d_in[5];
    const float* bn2_b   = (const float*)d_in[6];
    const float* le0_w1  = (const float*)d_in[7];
    const float* le0_g1  = (const float*)d_in[8];
    const float* le0_b1  = (const float*)d_in[9];
    const float* le0_w2  = (const float*)d_in[10];
    const float* le0_g2  = (const float*)d_in[11];
    const float* le0_b2  = (const float*)d_in[12];
    const float* le1_w1  = (const float*)d_in[13];
    const float* le1_g1  = (const float*)d_in[14];
    const float* le1_b1  = (const float*)d_in[15];
    const float* le1_w2  = (const float*)d_in[16];
    const float* le1_g2  = (const float*)d_in[17];
    const float* le1_b2  = (const float*)d_in[18];
    float* f_out = (float*)d_out;

    char* p = (char*)d_ws;
    auto alloc = [&](size_t bytes) -> void* {
        void* r = (void*)p;
        p += (bytes + 255) & ~(size_t)255;
        return r;
    };
    float*    y1    = (float*)alloc((size_t)B_ * C1_ * N_ * 4);
    float*    y1n   = (float*)alloc((size_t)B_ * C1_ * N_ * 4);
    float*    y2    = (float*)alloc((size_t)B_ * C1_ * N_ * 4);
    _Float16* ptsh  = (_Float16*)alloc((size_t)B_ * N_ * C1_ * 2);
    float*    xyzw  = (float*)alloc((size_t)B_ * N_ * 4 * 4);
    float*    sb1   = (float*)alloc(2 * C1_ * 4);
    float*    sb2   = (float*)alloc(2 * C1_ * 4);
    float*    sbE0a = (float*)alloc(2 * D0_ * 4);
    float*    sbE0b = (float*)alloc(2 * D0_ * 4);
    float*    sbE1a = (float*)alloc(2 * D1_ * 4);
    float*    sbE1b = (float*)alloc(2 * D1_ * 4);
    int*      fps0  = (int*)alloc((size_t)B_ * NP0_ * 4);
    float*    nxyz0 = (float*)alloc((size_t)B_ * NP0_ * 4 * 4);
    int*      knn0  = (int*)alloc((size_t)B_ * NP0_ * NS_ * 4);
    _Float16* pts1h = (_Float16*)alloc((size_t)B_ * NP0_ * D0_ * 2);
    int*      fps1  = (int*)alloc((size_t)B_ * NP1_ * 4);
    float*    nxyz1 = (float*)alloc((size_t)B_ * NP1_ * 4 * 4);
    int*      knn1  = (int*)alloc((size_t)B_ * NP1_ * NS_ * 4);
    _Float16* w0a   = (_Float16*)alloc((size_t)D0_ * D0_ * 2);
    _Float16* w0b   = (_Float16*)alloc((size_t)D0_ * D0_ * 2);
    _Float16* w1a   = (_Float16*)alloc((size_t)D1_ * D1_ * 2);
    _Float16* w1b   = (_Float16*)alloc((size_t)D1_ * D1_ * 2);
    _Float16* bufA  = (_Float16*)alloc((size_t)B_ * NP0_ * D0_ * NS_ * 2); // == B*NP1*D1*NS
    _Float16* bufB  = (_Float16*)alloc((size_t)B_ * NP0_ * D0_ * NS_ * 2);

    // ---- conv stack ----
    conv1_kernel<<<dim3(N_ / 256, C1_, B_), 256, 0, stream>>>(x, conv1_w, y1);
    conv_stats_kernel<<<C1_, 256, 0, stream>>>(y1, bn1_g, bn1_b, sb1, C1_, N_, B_);
    bnrelu_kernel<<<(B_ * C1_ * N_) / 256, 256, 0, stream>>>(y1, sb1, y1n, B_ * C1_ * N_);
    conv2_kernel<<<dim3(N_ / 256, C1_, B_), 256, 0, stream>>>(y1n, conv2_w, y2);
    conv_stats_kernel<<<C1_, 256, 0, stream>>>(y2, bn2_g, bn2_b, sb2, C1_, N_, B_);
    pts_kernel<<<(B_ * N_ * C1_) / 256, 256, 0, stream>>>(y2, sb2, ptsh, B_ * N_ * C1_);
    xyz_kernel<<<(B_ * N_) / 256, 256, 0, stream>>>(x, xyzw);

    // ---- weight conversion ----
    cvt_kernel<<<(D0_ * D0_) / 256, 256, 0, stream>>>(le0_w1, w0a, D0_ * D0_);
    cvt_kernel<<<(D0_ * D0_) / 256, 256, 0, stream>>>(le0_w2, w0b, D0_ * D0_);
    cvt_kernel<<<(D1_ * D1_) / 256, 256, 0, stream>>>(le1_w1, w1a, D1_ * D1_);
    cvt_kernel<<<(D1_ * D1_) / 256, 256, 0, stream>>>(le1_w2, w1b, D1_ * D1_);

    // ---- sample & group (stage 0) ----
    fps_kernel<N_, NP0_, 512><<<B_, 512, 0, stream>>>(xyzw, fps0);
    gatherxyz_kernel<<<(B_ * NP0_) / 256, 256, 0, stream>>>(xyzw, fps0, nxyz0,
                                                            NP0_, N_, B_ * NP0_);
    knn_kernel<N_, 256><<<B_ * NP0_, 256, 0, stream>>>(nxyz0, xyzw, knn0, NP0_);

    // ---- local encoder 0 (D=128) ----
    enc_gemm1_kernel<D0_><<<B_ * NP0_, 256, 0, stream>>>(ptsh, N_, fps0, knn0,
                                                         w0a, bufA, NP0_);
    enc_stats_kernel<<<D0_, 256, 0, stream>>>(bufA, D0_, B_ * NP0_, le0_g1, le0_b1, sbE0a);
    enc_gemm2_kernel<D0_><<<B_ * NP0_, 256, 0, stream>>>(bufA, sbE0a, w0b, bufB);
    enc_stats_kernel<<<D0_, 256, 0, stream>>>(bufB, D0_, B_ * NP0_, le0_g2, le0_b2, sbE0b);
    enc_final_kernel<D0_, false><<<(B_ * NP0_ * D0_) / 256, 256, 0, stream>>>(
        bufB, sbE0b, pts1h, nullptr, NP0_, B_ * NP0_ * D0_);

    // ---- sample & group (stage 1) ----
    fps_kernel<NP0_, NP1_, 256><<<B_, 256, 0, stream>>>(nxyz0, fps1);
    gatherxyz_kernel<<<(B_ * NP1_) / 256, 256, 0, stream>>>(nxyz0, fps1, nxyz1,
                                                            NP1_, NP0_, B_ * NP1_);
    knn_kernel<NP0_, 256><<<B_ * NP1_, 256, 0, stream>>>(nxyz1, nxyz0, knn1, NP1_);

    // ---- local encoder 1 (D=256) ----
    enc_gemm1_kernel<D1_><<<B_ * NP1_, 256, 0, stream>>>(pts1h, NP0_, fps1, knn1,
                                                         w1a, bufA, NP1_);
    enc_stats_kernel<<<D1_, 256, 0, stream>>>(bufA, D1_, B_ * NP1_, le1_g1, le1_b1, sbE1a);
    enc_gemm2_kernel<D1_><<<B_ * NP1_, 256, 0, stream>>>(bufA, sbE1a, w1b, bufB);
    enc_stats_kernel<<<D1_, 256, 0, stream>>>(bufB, D1_, B_ * NP1_, le1_g2, le1_b2, sbE1b);
    enc_final_kernel<D1_, true><<<(B_ * NP1_ * D1_) / 256, 256, 0, stream>>>(
        bufB, sbE1b, nullptr, f_out, NP1_, B_ * NP1_ * D1_);
}
